// CausalSelfAttention_38878043964081
// MI455X (gfx1250) — matrix operand order repaired and checked by reading, not compiled, not run on hardware
//
#include <hip/hip_runtime.h>
#include <hip/hip_bf16.h>

typedef __attribute__((ext_vector_type(16))) _Float16 v16h;
typedef __attribute__((ext_vector_type(8)))  _Float16 v8h;
typedef __attribute__((ext_vector_type(8)))  float    v8f;

#define BB 8
#define TT 1024
#define CC 768
#define NH 12
#define HD 64
#define QKVC (3 * CC)
#define NEGBIG (-3.0e38f)

static __device__ inline v16h cat16(v8h lo, v8h hi) {
    return __builtin_shufflevector(lo, hi, 0, 1, 2, 3, 4, 5, 6, 7, 8, 9, 10, 11, 12, 13, 14, 15);
}

static __device__ inline v8f wmma32(v16h a, v16h b, v8f c) {
    // D = A(16x32 f16) * B(32x16 f16) + C(16x16 f32)
    return __builtin_amdgcn_wmma_f32_16x16x32_f16(false, a, false, b, (short)0, c, false, false);
}

// ---------------------------------------------------------------------------
// Tensor Data Mover: 2D tile (global, strided) -> LDS with row padding.
// D# packing per CDNA5 ISA. 6-arg builtin form.
// ---------------------------------------------------------------------------
#if __has_builtin(__builtin_amdgcn_tensor_load_to_lds) && __has_builtin(__builtin_amdgcn_s_wait_tensorcnt)
#define HAVE_TDM 1
typedef __attribute__((ext_vector_type(4))) unsigned int u32x4;
typedef __attribute__((ext_vector_type(8))) int          i32x8;
typedef __attribute__((ext_vector_type(4))) int          i32x4;

static __device__ inline void tdm_load_2d(unsigned lds_off, const void* gptr,
                                          unsigned tile_w /*f16 elems*/, unsigned tile_h /*rows*/,
                                          unsigned src_stride /*f16 elems*/,
                                          unsigned pad_interval_code, unsigned pad_amount_code) {
    unsigned long long ga = (unsigned long long)(uintptr_t)gptr;
    u32x4 g0;
    g0[0] = 1u;                                               // count=1, user mode
    g0[1] = lds_off;                                          // lds_addr
    g0[2] = (unsigned)(ga & 0xffffffffu);                     // global_addr[31:0]
    g0[3] = (unsigned)((ga >> 32) & 0x1ffffffu) | (2u << 30); // global_addr[56:32] | type=2
    i32x8 g1;
    unsigned w0 = (1u << 16)                                  // data_size = 2 bytes
                | (1u << 20)                                  // pad_enable
                | (pad_interval_code << 22)
                | (pad_amount_code << 25);
    g1[0] = (int)w0;
    g1[1] = (int)(tile_w << 16);                              // tensor_dim0[15:0]
    g1[2] = (int)(tile_h << 16);                              // tensor_dim1[15:0]
    g1[3] = (int)(tile_w << 16);                              // tile_dim0
    g1[4] = (int)tile_h;                                      // tile_dim1 (tile_dim2 = 0)
    g1[5] = (int)src_stride;                                  // tensor_dim0_stride[31:0]
    g1[6] = 0;
    g1[7] = 0;
    i32x4 z4 = {0, 0, 0, 0};
    i32x8 z8 = {0, 0, 0, 0, 0, 0, 0, 0};
    __builtin_amdgcn_tensor_load_to_lds(g0, g1, z4, z4, z8, 0);
}
#else
#define HAVE_TDM 0
#endif

// ---------------------------------------------------------------------------
// Converters
// ---------------------------------------------------------------------------
__global__ void cvt_f16(const float* __restrict__ in, _Float16* __restrict__ out, int n) {
    int i = blockIdx.x * blockDim.x + threadIdx.x;
    if (i < n) out[i] = (_Float16)in[i];
}

// in: [K][N] f32  ->  out: [N][K] f16
__global__ void cvt_f16_T(const float* __restrict__ in, _Float16* __restrict__ out, int K, int N) {
    int i = blockIdx.x * blockDim.x + threadIdx.x;
    if (i < K * N) {
        int k = i / N, n = i % N;
        out[(size_t)n * K + k] = (_Float16)in[i];
    }
}

// ---------------------------------------------------------------------------
// GEMM: Out[M][N] = A[M][K](f16) * BT[N][K]^T(f16) + bias[N](f32)
// block tile 64x128, 8 waves (2x4), wave tile 32x32, K-step 32.
// Double-buffered LDS: TDM for tile i+1 overlaps WMMA on tile i.
// ---------------------------------------------------------------------------
template <bool OUT_F16>
__global__ __launch_bounds__(256) void gemm_f16(const _Float16* __restrict__ A,
                                                const _Float16* __restrict__ BT,
                                                const float* __restrict__ bias,
                                                void* __restrict__ Out,
                                                int M, int N, int K) {
    __shared__ __align__(16) _Float16 lA[2][64][40];    // 64x32 + 8 pad (80B stride)
    __shared__ __align__(16) _Float16 lB[2][128][40];   // 128x32 + 8 pad (N-major)

    const int tid  = threadIdx.x;
    const int lane = tid & 31;
    const int w    = tid >> 5;
    const int wr   = w >> 2;
    const int wc   = w & 3;
    const int lm   = lane & 15;
    const int lh   = lane >> 4;
    const int mb   = blockIdx.y * 64;
    const int nb   = blockIdx.x * 128;

    auto stage = [&](int buf, int k0) {
#if HAVE_TDM
        if (w == 0) {
            tdm_load_2d((unsigned)(uintptr_t)&lA[buf][0][0], A + (size_t)mb * K + k0,
                        32, 64, (unsigned)K, /*64B rows*/ 3, /*+16B pad*/ 3);
            tdm_load_2d((unsigned)(uintptr_t)&lB[buf][0][0], BT + (size_t)nb * K + k0,
                        32, 128, (unsigned)K, 3, 3);
        }
#else
        {
            int row = tid >> 2;
            int col = (tid & 3) * 8;
            *(v8h*)&lA[buf][row][col] = *(const v8h*)(A + (size_t)(mb + row) * K + k0 + col);
        }
        {
            int row = tid >> 1;
            int col = (tid & 1) * 16;
            const v8h* src = (const v8h*)(BT + (size_t)(nb + row) * K + k0 + col);
            *(v8h*)&lB[buf][row][col]     = src[0];
            *(v8h*)&lB[buf][row][col + 8] = src[1];
        }
#endif
    };

    v8f c00 = {}, c01 = {}, c10 = {}, c11 = {};
    const int nsteps = K / 32;

    stage(0, 0);
    for (int i = 0; i < nsteps; ++i) {
        const int cur = i & 1;
        const int k0  = i * 32;
#if HAVE_TDM
        if (w == 0) __builtin_amdgcn_s_wait_tensorcnt((short)0);
#endif
        __syncthreads();                       // tile[cur] ready; tile[1-cur] free
        if (i + 1 < nsteps) stage(1 - cur, k0 + 32);
        if (i + 2 < nsteps) {                  // global_prefetch_b8 two tiles ahead
            __builtin_prefetch(A + (size_t)(mb + (tid >> 2)) * K + k0 + 64 + (tid & 3) * 8, 0, 1);
            __builtin_prefetch(BT + (size_t)(nb + (tid >> 1)) * K + k0 + 64 + (tid & 1) * 16, 0, 1);
        }

        const int am = wr * 32 + lm;
        const int ko = lh * 8;
        v16h a0 = cat16(*(const v8h*)&lA[cur][am][ko],      *(const v8h*)&lA[cur][am][16 + ko]);
        v16h a1 = cat16(*(const v8h*)&lA[cur][am + 16][ko], *(const v8h*)&lA[cur][am + 16][16 + ko]);

        const int bn = wc * 32 + lm;
        const int kb = lh * 8;
        v16h b0 = cat16(*(const v8h*)&lB[cur][bn][kb],      *(const v8h*)&lB[cur][bn][kb + 16]);
        v16h b1 = cat16(*(const v8h*)&lB[cur][bn + 16][kb], *(const v8h*)&lB[cur][bn + 16][kb + 16]);

        c00 = wmma32(a0, b0, c00);
        c01 = wmma32(a0, b1, c01);
        c10 = wmma32(a1, b0, c10);
        c11 = wmma32(a1, b1, c11);
    }

    const int mBase = mb + wr * 32;
    const int nBase = nb + wc * 32;
    v8f cc[2][2] = {{c00, c01}, {c10, c11}};
#pragma unroll
    for (int i = 0; i < 2; ++i) {
#pragma unroll
        for (int j = 0; j < 2; ++j) {
            int colg = nBase + j * 16 + lm;
            float bv = bias[colg];
#pragma unroll
            for (int r = 0; r < 8; ++r) {
                int rowg  = mBase + i * 16 + r + 8 * lh;
                float val = cc[i][j][r] + bv;
                if constexpr (OUT_F16)
                    ((_Float16*)Out)[(size_t)rowg * N + colg] = (_Float16)val;
                else
                    ((float*)Out)[(size_t)rowg * N + colg] = val;
            }
        }
    }
}

// ---------------------------------------------------------------------------
// Attention: flash-style, 64-key chunks, double-buffered K (TDM) and V^T
// (manual transpose) staging. 128 threads = 4 waves; each wave owns a 16-row
// q tile. Row-sum via ones-WMMA; only row-max needs cross-lane shuffles.
// ---------------------------------------------------------------------------
__global__ __launch_bounds__(128) void attn_kernel(const _Float16* __restrict__ qkv,
                                                   const int* __restrict__ mask,
                                                   _Float16* __restrict__ out) {
    __shared__ __align__(16) _Float16 lK[2][64][72];     // 64 keys x 64 d (+8 pad)
    __shared__ __align__(16) _Float16 lVt[2][HD][72];    // V^T: [d][64 keys (+8 pad)]
    __shared__ __align__(16) _Float16 lP[4][16][72];     // per-wave P (16x64) scratch

    const int tid  = threadIdx.x;
    const int lane = tid & 31;
    const int w    = tid >> 5;    // 0..3
    const int lm   = lane & 15;
    const int lh   = lane >> 4;

    const int bid = blockIdx.x;
    const int qt  = bid & 15;
    const int h   = (bid >> 4) % NH;
    const int b   = bid / (16 * NH);
    const size_t tokbase = (size_t)b * TT;

    auto stageK = [&](int buf, int kv0) {
#if HAVE_TDM
        if (w == 0) {   // K tile 64x64 halves, stride QKVC, 128B rows + 16B pad
            tdm_load_2d((unsigned)(uintptr_t)&lK[buf][0][0],
                        qkv + (tokbase + kv0) * QKVC + CC + h * HD,
                        64, 64, QKVC, /*128B rows*/ 4, /*+16B pad*/ 3);
        }
#else
        int row = tid >> 1;
        int col = (tid & 1) * 32;
        const _Float16* kp = qkv + (tokbase + kv0 + row) * QKVC + CC + h * HD + col;
        *(v8h*)&lK[buf][row][col]      = *(const v8h*)kp;
        *(v8h*)&lK[buf][row][col + 8]  = *(const v8h*)(kp + 8);
        *(v8h*)&lK[buf][row][col + 16] = *(const v8h*)(kp + 16);
        *(v8h*)&lK[buf][row][col + 24] = *(const v8h*)(kp + 24);
#endif
    };

    auto stageV = [&](int buf, int kv0) {   // transposed; TDM cannot transpose
        int row = tid >> 1;            // key row 0..63
        int col = (tid & 1) * 32;      // d base
        const _Float16* vp = qkv + (tokbase + kv0 + row) * QKVC + 2 * CC + h * HD + col;
        v8h v0 = *(const v8h*)vp;
        v8h v1 = *(const v8h*)(vp + 8);
        v8h v2 = *(const v8h*)(vp + 16);
        v8h v3 = *(const v8h*)(vp + 24);
#pragma unroll
        for (int i = 0; i < 8; ++i) {
            lVt[buf][col + i][row]      = v0[i];
            lVt[buf][col + 8 + i][row]  = v1[i];
            lVt[buf][col + 16 + i][row] = v2[i];
            lVt[buf][col + 24 + i][row] = v3[i];
        }
    };

    // Q fragments, pre-scaled by 1/sqrt(hd)
    const int qrow = qt * 64 + w * 16 + lm;
    const _Float16* qptr = qkv + (tokbase + qrow) * QKVC + h * HD;
    v16h qf[2];
#pragma unroll
    for (int f = 0; f < 2; ++f) {
        int d0 = f * 32 + lh * 8;
        v16h q = cat16(*(const v8h*)(qptr + d0), *(const v8h*)(qptr + d0 + 16));
        qf[f]  = q * (_Float16)0.125f;
    }

    v16h ones;
#pragma unroll
    for (int i = 0; i < 16; ++i) ones[i] = (_Float16)1.0f;

    float mi[8];
    v8f o[4] = {};
    v8f lacc = {};     // running softmax denominator (every lane holds its row's sum)
#pragma unroll
    for (int r = 0; r < 8; ++r) mi[r] = NEGBIG;

    const int nchunks = TT / 64;
    stageK(0, 0);
    stageV(0, 0);
    for (int i = 0; i < nchunks; ++i) {
        const int cur = i & 1;
        const int kv0 = i * 64;
#if HAVE_TDM
        if (w == 0) __builtin_amdgcn_s_wait_tensorcnt((short)0);
#endif
        __syncthreads();                       // chunk[cur] ready; chunk[1-cur] free
        if (i + 1 < nchunks) {
            stageK(1 - cur, kv0 + 64);
            stageV(1 - cur, kv0 + 64);
        }

        // S = Q * K^T for four 16-key tiles
        v8f s[4] = {};
#pragma unroll
        for (int kt = 0; kt < 4; ++kt) {
#pragma unroll
            for (int f = 0; f < 2; ++f) {
                int dof = f * 32 + lh * 8;
                v16h bk = cat16(*(const v8h*)&lK[cur][kt * 16 + lm][dof],
                                *(const v8h*)&lK[cur][kt * 16 + lm][dof + 16]);
                s[kt] = wmma32(qf[f], bk, s[kt]);
            }
        }

        bool ok0 = mask[b * TT + kv0 + lm] != 0;
        bool ok1 = mask[b * TT + kv0 + 16 + lm] != 0;
        bool ok2 = mask[b * TT + kv0 + 32 + lm] != 0;
        bool ok3 = mask[b * TT + kv0 + 48 + lm] != 0;

#pragma unroll
        for (int r = 0; r < 8; ++r) {
            float e0 = ok0 ? s[0][r] : NEGBIG;
            float e1 = ok1 ? s[1][r] : NEGBIG;
            float e2 = ok2 ? s[2][r] : NEGBIG;
            float e3 = ok3 ? s[3][r] : NEGBIG;
            float m  = fmaxf(fmaxf(e0, e1), fmaxf(e2, e3));
#pragma unroll
            for (int d = 1; d < 16; d <<= 1) m = fmaxf(m, __shfl_xor(m, d, 16));
            float mn = fmaxf(mi[r], m);
            float al = __expf(mi[r] - mn);
            mi[r]    = mn;
            float p0 = __expf(e0 - mn);
            float p1 = __expf(e1 - mn);
            float p2 = __expf(e2 - mn);
            float p3 = __expf(e3 - mn);
            o[0][r] *= al; o[1][r] *= al; o[2][r] *= al; o[3][r] *= al;
            lacc[r] *= al;
            int prow = r + 8 * lh;
            lP[w][prow][lm]      = (_Float16)p0;
            lP[w][prow][16 + lm] = (_Float16)p1;
            lP[w][prow][32 + lm] = (_Float16)p2;
            lP[w][prow][48 + lm] = (_Float16)p3;
        }

        // P (16x64) as two A fragments; O += P*V, lacc += P*ones (row sums)
        {
            int ko = lh * 8;
            v16h pf0 = cat16(*(const v8h*)&lP[w][lm][ko],      *(const v8h*)&lP[w][lm][16 + ko]);
            v16h pf1 = cat16(*(const v8h*)&lP[w][lm][32 + ko], *(const v8h*)&lP[w][lm][48 + ko]);
            lacc = wmma32(pf0, ones, lacc);
            lacc = wmma32(pf1, ones, lacc);
            int kc = lh * 8;
#pragma unroll
            for (int t = 0; t < 4; ++t) {
                v16h vf0 = cat16(*(const v8h*)&lVt[cur][t * 16 + lm][kc],
                                 *(const v8h*)&lVt[cur][t * 16 + lm][kc + 16]);
                v16h vf1 = cat16(*(const v8h*)&lVt[cur][t * 16 + lm][32 + kc],
                                 *(const v8h*)&lVt[cur][t * 16 + lm][32 + kc + 16]);
                o[t] = wmma32(pf0, vf0, o[t]);
                o[t] = wmma32(pf1, vf1, o[t]);
            }
        }
    }

    // normalize (denominator already in every lane) and store f16 [B*T][C]
    const int orow = qt * 64 + w * 16;
#pragma unroll
    for (int r = 0; r < 8; ++r) {
        float inv = 1.0f / lacc[r];
        int row   = orow + r + 8 * lh;
        size_t rb = (tokbase + row) * CC + h * HD;
#pragma unroll
        for (int t = 0; t < 4; ++t)
            out[rb + t * 16 + lm] = (_Float16)(o[t][r] * inv);
    }
}

// ---------------------------------------------------------------------------
extern "C" void kernel_launch(void* const* d_in, const int* in_sizes, int n_in,
                              void* d_out, int out_size, void* d_ws, size_t ws_size,
                              hipStream_t stream) {
    const float* x      = (const float*)d_in[0];
    const int*   amask  = (const int*)d_in[1];
    const float* W_attn = (const float*)d_in[2];
    const float* b_attn = (const float*)d_in[3];
    const float* W_proj = (const float*)d_in[4];
    const float* b_proj = (const float*)d_in[5];
    float*       outp   = (float*)d_out;

    const int M = BB * TT;  // 8192

    char* ws = (char*)d_ws;
    _Float16* xh   = (_Float16*)ws;   ws += (size_t)M * CC * 2;
    _Float16* WaT  = (_Float16*)ws;   ws += (size_t)QKVC * CC * 2;
    _Float16* WpT  = (_Float16*)ws;   ws += (size_t)CC * CC * 2;
    _Float16* qkvh = (_Float16*)ws;   ws += (size_t)M * QKVC * 2;
    _Float16* aoh  = (_Float16*)ws;

    cvt_f16<<<(M * CC + 255) / 256, 256, 0, stream>>>(x, xh, M * CC);
    cvt_f16_T<<<(CC * QKVC + 255) / 256, 256, 0, stream>>>(W_attn, WaT, CC, QKVC);
    cvt_f16_T<<<(CC * CC + 255) / 256, 256, 0, stream>>>(W_proj, WpT, CC, CC);

    gemm_f16<true><<<dim3(QKVC / 128, M / 64), 256, 0, stream>>>(
        xh, WaT, b_attn, (void*)qkvh, M, QKVC, CC);

    attn_kernel<<<BB * NH * (TT / 64), 128, 0, stream>>>(qkvh, amask, aoh);

    gemm_f16<false><<<dim3(CC / 128, M / 64), 256, 0, stream>>>(
        aoh, WpT, b_proj, (void*)outp, M, CC, CC);
}
